// ProgramStep_69406671503779
// MI455X (gfx1250) — matrix operand order repaired and checked by reading, not compiled, not run on hardware
//
#include <hip/hip_runtime.h>

typedef __attribute__((ext_vector_type(16))) _Float16 v16h;
typedef __attribute__((ext_vector_type(8)))  _Float16 v8h;
typedef __attribute__((ext_vector_type(8)))  float    v8f;

#define DIMN 32
#define KEXP 16
#define BATCH 262144
#define WAVES_PER_BLOCK 8
#define NBLOCKS 512
#define TOTAL_WAVES (NBLOCKS * WAVES_PER_BLOCK)          // 4096
#define NTILES (BATCH / 16)                              // 16384
#define TILES_PER_WAVE (NTILES / TOTAL_WAVES)            // 4

__global__ __launch_bounds__(256, 4) void moe_gated_kernel(
    const float* __restrict__ x,        // [B,32]
    const float* __restrict__ mu,       // [16,32]
    const float* __restrict__ logsigma, // [16,32]
    const float* __restrict__ W,        // [16,32,32]
    const float* __restrict__ bias,     // [16,32]
    float* __restrict__ out)            // [B,32]
{
    __shared__ __align__(32) _Float16 Wl[KEXP * DIMN * DIMN]; // 32 KB, [k][o][d] f16
    __shared__ __align__(32) _Float16 Pl[KEXP * DIMN];        // gate coeff P=-0.5/s2
    __shared__ __align__(32) _Float16 Ql[KEXP * DIMN];        // gate coeff Q=mu/s2
    __shared__ __align__(32) float Gc[KEXP];                  // gate constant per k
    __shared__ __align__(32) float Bl[KEXP * DIMN];           // bias f32
    __shared__ __align__(32) float wbuf[WAVES_PER_BLOCK][KEXP][16]; // w[k][m] per wave

    const int tid = threadIdx.x;
    if (tid < KEXP) Gc[tid] = 0.0f;
    __syncthreads();

    // W f32 -> f16 into LDS.
    for (int i = tid * 4; i < KEXP * DIMN * DIMN; i += 256 * 4) {
        const float4 wv = *(const float4*)(W + i);
        Wl[i + 0] = (_Float16)wv.x;
        Wl[i + 1] = (_Float16)wv.y;
        Wl[i + 2] = (_Float16)wv.z;
        Wl[i + 3] = (_Float16)wv.w;
    }
    // Gate coefficients: logps[k](x) = sum_d x^2*P + x*Q + C
    for (int i = tid; i < KEXP * DIMN; i += 256) {
        const int k = i / DIMN;
        const float ls = logsigma[i];
        const float s2inv = __expf(-2.0f * ls);
        const float m = mu[i];
        Pl[i] = (_Float16)(-0.5f * s2inv);
        Ql[i] = (_Float16)(m * s2inv);
        Bl[i] = bias[i];
        const float c = -0.5f * m * m * s2inv - ls - 0.9189385332046727f;
        atomicAdd(&Gc[k], c);   // ds_add_f32
    }
    __syncthreads();

    const int lane = tid & 31;
    const int wave = tid >> 5;
    const int col  = lane & 15;
    const int half = lane >> 4;

    // Gate A-fragments (rows = experts): lane holds G[col][h*16..+16]
    const v8h p0 = *(const v8h*)(&Pl[col * DIMN + half * 16]);
    const v8h p1 = *(const v8h*)(&Pl[col * DIMN + half * 16 + 8]);
    const v8h q0 = *(const v8h*)(&Ql[col * DIMN + half * 16]);
    const v8h q1 = *(const v8h*)(&Ql[col * DIMN + half * 16 + 8]);
    const v16h Pa = __builtin_shufflevector(p0, p1, 0,1,2,3,4,5,6,7,8,9,10,11,12,13,14,15);
    const v16h Qa = __builtin_shufflevector(q0, q1, 0,1,2,3,4,5,6,7,8,9,10,11,12,13,14,15);

    // Gate constant in C-operand layout: lane l, VGPR r -> k = half*8 + r.
    v8f gcreg;
#pragma unroll
    for (int r = 0; r < 8; ++r) gcreg[r] = Gc[half * 8 + r];

    const v8f czero = {};
    float* const wb = &wbuf[wave][0][0];
    // Loop-invariant bases for the rolled expert loop.
    const _Float16* const wp0 = &Wl[col * DIMN + half * 16];
    const _Float16* const wp1 = &Wl[(16 + col) * DIMN + half * 16];
    const float* const blp = &Bl[col];
    const float* const wvp = wb + half * 8;

    const int wgid = blockIdx.x * WAVES_PER_BLOCK + wave;

#pragma unroll 1
    for (int t = 0; t < TILES_PER_WAVE; ++t) {
        const int tile = wgid + t * TOTAL_WAVES;
        const int row0 = tile * 16;
        const float* xrow = x + (size_t)(row0 + col) * DIMN;

        // A-fragment for expert matmul: row col, dims h*16..+16
        const float4 a0 = *(const float4*)(xrow + half * 16);
        const float4 a1 = *(const float4*)(xrow + half * 16 + 4);
        const float4 a2 = *(const float4*)(xrow + half * 16 + 8);
        const float4 a3 = *(const float4*)(xrow + half * 16 + 12);
        v16h A;
        A[0]=(_Float16)a0.x;  A[1]=(_Float16)a0.y;  A[2]=(_Float16)a0.z;  A[3]=(_Float16)a0.w;
        A[4]=(_Float16)a1.x;  A[5]=(_Float16)a1.y;  A[6]=(_Float16)a1.z;  A[7]=(_Float16)a1.w;
        A[8]=(_Float16)a2.x;  A[9]=(_Float16)a2.y;  A[10]=(_Float16)a2.z; A[11]=(_Float16)a2.w;
        A[12]=(_Float16)a3.x; A[13]=(_Float16)a3.y; A[14]=(_Float16)a3.z; A[15]=(_Float16)a3.w;

        // B-fragment of x^T for transposed gating: col m, K-dims h*16..+16
        const float4 b0_ = *(const float4*)(xrow + half * 16);
        const float4 b1_ = *(const float4*)(xrow + half * 16 + 4);
        const float4 b2_ = *(const float4*)(xrow + half * 16 + 8);
        const float4 b3_ = *(const float4*)(xrow + half * 16 + 12);
        v16h Bx;
        Bx[0]=(_Float16)b0_.x;  Bx[1]=(_Float16)b0_.y;  Bx[2]=(_Float16)b0_.z;  Bx[3]=(_Float16)b0_.w;
        Bx[4]=(_Float16)b1_.x;  Bx[5]=(_Float16)b1_.y;  Bx[6]=(_Float16)b1_.z;  Bx[7]=(_Float16)b1_.w;
        Bx[8]=(_Float16)b2_.x;  Bx[9]=(_Float16)b2_.y;  Bx[10]=(_Float16)b2_.z; Bx[11]=(_Float16)b2_.w;
        Bx[12]=(_Float16)b3_.x; Bx[13]=(_Float16)b3_.y; Bx[14]=(_Float16)b3_.z; Bx[15]=(_Float16)b3_.w;
        const v16h Bxsq = Bx * Bx;

        // logps^T[k, m]: D lane l -> m = col, VGPR r -> k = half*8 + r.
        v8f accg = __builtin_amdgcn_wmma_f32_16x16x32_f16(false, Pa, false, Bxsq,
                                                          (short)0, gcreg, false, false);
        accg = __builtin_amdgcn_wmma_f32_16x16x32_f16(false, Qa, false, Bx,
                                                      (short)0, accg, false, false);

        // Softmax over k: 8 in-register + 1 cross-half exchange.
        float mx = accg[0];
#pragma unroll
        for (int r = 1; r < 8; ++r) mx = fmaxf(mx, accg[r]);
        mx = fmaxf(mx, __shfl_xor(mx, 16, 32));
        float e[8];
        float s = 0.0f;
#pragma unroll
        for (int r = 0; r < 8; ++r) { e[r] = __expf(accg[r] - mx); s += e[r]; }
        s += __shfl_xor(s, 16, 32);
        const float inv = 1.0f / s;

        // w[k][m] -> LDS (per-wave slab), read back vectorized in expert loop.
#pragma unroll
        for (int r = 0; r < 8; ++r) wb[(half * 8 + r) * 16 + col] = e[r] * inv;

        // ---- Experts (rolled: low VGPR pressure, no spills) ----
        v8f o0 = {}, o1 = {};
#pragma unroll 2
        for (int k = 0; k < KEXP; ++k) {
            const v16h B0 = *(const v16h*)(wp0 + k * (DIMN * DIMN));
            const v16h B1 = *(const v16h*)(wp1 + k * (DIMN * DIMN));
            const v8f wv = *(const v8f*)(wvp + k * 16);   // w[k][h*8+r], LDS broadcast
            const float bb0 = blp[k * DIMN];
            const float bb1 = blp[k * DIMN + 16];
            const v8f l0 = __builtin_amdgcn_wmma_f32_16x16x32_f16(false, A, false, B0,
                                                                  (short)0, czero, false, false);
            const v8f l1 = __builtin_amdgcn_wmma_f32_16x16x32_f16(false, A, false, B1,
                                                                  (short)0, czero, false, false);
#pragma unroll
            for (int r = 0; r < 8; ++r) {
                o0[r] = fmaf(wv[r], fmaxf(l0[r] + bb0, 0.0f), o0[r]);
                o1[r] = fmaf(wv[r], fmaxf(l1[r] + bb1, 0.0f), o1[r]);
            }
        }

        // Store: lane l -> rows (half*8 + r), cols {col, 16+col}
        float* ob = out + (size_t)(row0 + half * 8) * DIMN;
#pragma unroll
        for (int r = 0; r < 8; ++r) {
            ob[r * DIMN + col]      = o0[r];
            ob[r * DIMN + 16 + col] = o1[r];
        }
    }
}

extern "C" void kernel_launch(void* const* d_in, const int* in_sizes, int n_in,
                              void* d_out, int out_size, void* d_ws, size_t ws_size,
                              hipStream_t stream) {
    const float* x        = (const float*)d_in[0];
    const float* mu       = (const float*)d_in[1];
    const float* logsigma = (const float*)d_in[2];
    const float* W        = (const float*)d_in[3];
    const float* b        = (const float*)d_in[4];
    float* out            = (float*)d_out;
    moe_gated_kernel<<<dim3(NBLOCKS), dim3(256), 0, stream>>>(x, mu, logsigma, W, b, out);
}
